// DecisionMetaMamba_33105607917918
// MI455X (gfx1250) — matrix operand order repaired and checked by reading, not compiled, not run on hardware
//
#include <hip/hip_runtime.h>
#include <math.h>

// ---------------------------------------------------------------------------
// DecisionMetaMamba on gfx1250 (MI455X).
// GEMMs: f16 operands, f32 accumulate via V_WMMA_F32_16X16X32_F16.
// A tiles staged to LDS via the CDNA5 async engine (GLOBAL_LOAD_ASYNC_TO_LDS,
// ASYNCcnt) when the toolchain exposes it; ds_store fallback otherwise.
// Each wave computes a 2x2 grid of 16x16 tiles (4 wmma per fragment set).
// Residual stream / LN stats / scan recurrence / head stay fp32.
// ---------------------------------------------------------------------------

typedef __attribute__((ext_vector_type(16))) _Float16 v16h;
typedef __attribute__((ext_vector_type(8)))  _Float16 h8;
typedef __attribute__((ext_vector_type(8)))  float    v8f;
typedef int v4i __attribute__((vector_size(16)));

#define HDIM   256
#define NLAYERS 3
#define WWIN   6
#define DSTATE 16
#define DCONV  4
#define DTRANK 16
#define SDIM   17
#define ADIM   6
#define BATCH  32
#define TSEQ   1024
#define LSEQ   (3*TSEQ)          /* 3072 */
#define MTOT   (BATCH*TSEQ*3)    /* 98304 */
#define EPSF   1e-5f

// ---- CDNA5 async global->LDS copy (probe-guarded) --------------------------
#define AS1 __attribute__((address_space(1)))
#define AS3 __attribute__((address_space(3)))
#if defined(__has_builtin)
#if __has_builtin(__builtin_amdgcn_global_load_async_to_lds_b128) && \
    __has_builtin(__builtin_amdgcn_s_wait_asynccnt)
#define HAVE_ASYNC_LDS 1
#endif
#endif
#ifndef HAVE_ASYNC_LDS
#define HAVE_ASYNC_LDS 0
#endif

__device__ __forceinline__ void async_copy16(const _Float16* g, _Float16* l)
{
#if HAVE_ASYNC_LDS
    __builtin_amdgcn_global_load_async_to_lds_b128(
        (AS1 v4i*)g, (AS3 v4i*)l, 0, 0);
#else
    *(h8*)l = *(const h8*)g;
#endif
}
__device__ __forceinline__ void async_join()
{
#if HAVE_ASYNC_LDS
    __builtin_amdgcn_s_wait_asynccnt(0);
#endif
}

__device__ __forceinline__ float sigmoidf_(float x){ return 1.0f/(1.0f+__expf(-x)); }
__device__ __forceinline__ float siluf_(float x){ return x*sigmoidf_(x); }
__device__ __forceinline__ float softplusf_(float x){ return (x>20.0f)? x : log1pf(__expf(x)); }

// ---------------------------------------------------------------------------
// elementwise f32 -> f16 (weight conversion, once per launch)
// ---------------------------------------------------------------------------
__global__ void f32_to_f16_kernel(const float* __restrict__ src,
                                  _Float16* __restrict__ dst, int n)
{
    int i = blockIdx.x*256 + threadIdx.x;
    if (i < n) dst[i] = (_Float16)src[i];
}

// ---------------------------------------------------------------------------
// Embedding (returns/state/action interleave) + LayerNorm -> hs (f32)
// ---------------------------------------------------------------------------
__global__ void embed_ln_kernel(const float* __restrict__ states,
                                const float* __restrict__ actions,
                                const float* __restrict__ rtg,
                                const float* __restrict__ W_ret, const float* __restrict__ b_ret,
                                const float* __restrict__ W_st,  const float* __restrict__ b_st,
                                const float* __restrict__ W_act, const float* __restrict__ b_act,
                                const float* __restrict__ g,     const float* __restrict__ be,
                                float* __restrict__ hs)
{
    int tok = blockIdx.x;            // 0..MTOT-1
    int bb  = tok / LSEQ;
    int l   = tok % LSEQ;
    int t   = l / 3;
    int j   = l % 3;                 // 0=return, 1=state, 2=action
    int h   = threadIdx.x;

    float e;
    if (j == 0) {
        float r = rtg[bb*TSEQ + t];
        e = r * W_ret[h] + b_ret[h];
    } else if (j == 1) {
        const float* srow = states + (bb*TSEQ + t)*SDIM;
        const float* wrow = W_st + h*SDIM;
        float acc = b_st[h];
        #pragma unroll
        for (int k = 0; k < SDIM; k++) acc += srow[k]*wrow[k];
        e = acc;
    } else {
        const float* arow = actions + (bb*TSEQ + t)*ADIM;
        const float* wrow = W_act + h*ADIM;
        float acc = b_act[h];
        #pragma unroll
        for (int k = 0; k < ADIM; k++) acc += arow[k]*wrow[k];
        e = acc;
    }

    __shared__ float s1[HDIM], s2[HDIM];
    s1[h] = e; s2[h] = e*e;
    __syncthreads();
    for (int off = 128; off > 0; off >>= 1) {
        if (h < off) { s1[h] += s1[h+off]; s2[h] += s2[h+off]; }
        __syncthreads();
    }
    float mean = s1[0] * (1.0f/HDIM);
    float var  = s2[0] * (1.0f/HDIM) - mean*mean;
    float rstd = rsqrtf(var + EPSF);
    hs[(long)tok*HDIM + h] = (e - mean)*rstd*g[h] + be[h];
}

// ---------------------------------------------------------------------------
// LayerNorm with fused f16 downcast: out16 = (f16) (LN(in)*g + b)
// ---------------------------------------------------------------------------
__global__ void ln_f16_kernel(const float* __restrict__ in,
                              const float* __restrict__ g, const float* __restrict__ be,
                              _Float16* __restrict__ outp)
{
    int tok = blockIdx.x;
    int h   = threadIdx.x;
    float x = in[(long)tok*HDIM + h];
    __shared__ float s1[HDIM], s2[HDIM];
    s1[h] = x; s2[h] = x*x;
    __syncthreads();
    for (int off = 128; off > 0; off >>= 1) {
        if (h < off) { s1[h] += s1[h+off]; s2[h] += s2[h+off]; }
        __syncthreads();
    }
    float mean = s1[0] * (1.0f/HDIM);
    float var  = s2[0] * (1.0f/HDIM) - mean*mean;
    float rstd = rsqrtf(var + EPSF);
    outp[(long)tok*HDIM + h] = (_Float16)((x - mean)*rstd*g[h] + be[h]);
}

// ---------------------------------------------------------------------------
// Fragment helpers (ISA 16-bit layouts):
// A (16x32): lane(lr,hl): 16B chunks at k+8*hl and k+16+8*hl of row lr.
// B (32x16): lane(lr,hl): 16B chunks at k+8*hl and k+16+8*hl of W[n+lr].
// C/D (16x16 f32): lane<16 -> (M=v,N=lane); lane>=16 -> (M=v+8,N=lane-16).
// ---------------------------------------------------------------------------
union frag16 { v16h v; h8 h[2]; };

// ---------------------------------------------------------------------------
// f16 WMMA GEMM, async-LDS-staged A, 2x2 tiles per wave.
//   out[m,n] (=|+=) sum_k A[m,k]*Wt[n,k]
// block: 128 thr = 4 waves; covers 32 rows x 128 cols; grid=(M/32, ceil(N/128))
// ---------------------------------------------------------------------------
__global__ void gemm_f16_wmma(const _Float16* __restrict__ A, int lda,
                              const _Float16* __restrict__ Wt, int ldw, int K,
                              float* __restrict__ outf,      // f32 dest (or null)
                              _Float16* __restrict__ outh,   // f16 dest (or null)
                              int ldo, int N, int accumulate)
{
    __shared__ __align__(16) _Float16 sA[32*HDIM];

    int tid  = threadIdx.x;
    int lane = tid & 31;
    int wave = tid >> 5;
    int m0 = blockIdx.x * 32;
    int nb = blockIdx.y * 128 + wave*32;
    int hl = lane >> 4;
    int lr = lane & 15;

    // ---- stage A tile (32 x K halves) into LDS ----
    int nch = 32*(K/8);
    for (int ci = tid; ci < nch; ci += 128) {
        int row = ci / (K/8);
        int c8  = (ci % (K/8))*8;
        async_copy16(A + (long)(m0+row)*lda + c8, sA + row*HDIM + c8);
    }
    async_join();
    __syncthreads();

    bool v0 = (nb + 16) <= N;       // n-tile 0 valid
    bool v1 = (nb + 32) <= N;       // n-tile 1 valid
    if (!v0 && !v1) return;         // wave-uniform (after barrier)

    v8f c00, c01, c10, c11;
    #pragma unroll
    for (int v = 0; v < 8; v++) { c00[v]=0.f; c01[v]=0.f; c10[v]=0.f; c11[v]=0.f; }
    if (accumulate) {
        #pragma unroll
        for (int v = 0; v < 8; v++) {
            int r0 = m0 + v + 8*hl, r1 = r0 + 16;
            c00[v] = outf[(long)r0*ldo + nb      + lr];
            c10[v] = outf[(long)r1*ldo + nb      + lr];
            if (v1) {
                c01[v] = outf[(long)r0*ldo + nb + 16 + lr];
                c11[v] = outf[(long)r1*ldo + nb + 16 + lr];
            }
        }
    }

    const _Float16* a0 = sA + (lr     )*HDIM;
    const _Float16* a1 = sA + (lr + 16)*HDIM;
    const _Float16* w0 = Wt + (long)(nb      + lr)*ldw;
    const _Float16* w1 = Wt + (long)(nb + 16 + lr)*ldw;

    for (int k = 0; k < K; k += 32) {
        frag16 fa0, fa1, fb0, fb1;
        fa0.h[0] = *(const h8*)(a0 + k + 8*hl);
        fa0.h[1] = *(const h8*)(a0 + k + 16 + 8*hl);
        fa1.h[0] = *(const h8*)(a1 + k + 8*hl);
        fa1.h[1] = *(const h8*)(a1 + k + 16 + 8*hl);
        fb0.h[0] = *(const h8*)(w0 + k + 8*hl);
        fb0.h[1] = *(const h8*)(w0 + k + 16 + 8*hl);
        fb1.h[0] = *(const h8*)(w1 + k + 8*hl);
        fb1.h[1] = *(const h8*)(w1 + k + 16 + 8*hl);
        c00 = __builtin_amdgcn_wmma_f32_16x16x32_f16(false, fa0.v, false, fb0.v, (short)0, c00, false, false);
        c10 = __builtin_amdgcn_wmma_f32_16x16x32_f16(false, fa1.v, false, fb0.v, (short)0, c10, false, false);
        c01 = __builtin_amdgcn_wmma_f32_16x16x32_f16(false, fa0.v, false, fb1.v, (short)0, c01, false, false);
        c11 = __builtin_amdgcn_wmma_f32_16x16x32_f16(false, fa1.v, false, fb1.v, (short)0, c11, false, false);
    }

    #pragma unroll
    for (int v = 0; v < 8; v++) {
        int r0 = m0 + v + 8*hl, r1 = r0 + 16;
        if (outh) {
            if (v0) { outh[(long)r0*ldo + nb + lr]      = (_Float16)c00[v];
                      outh[(long)r1*ldo + nb + lr]      = (_Float16)c10[v]; }
            if (v1) { outh[(long)r0*ldo + nb + 16 + lr] = (_Float16)c01[v];
                      outh[(long)r1*ldo + nb + 16 + lr] = (_Float16)c11[v]; }
        } else {
            if (v0) { outf[(long)r0*ldo + nb + lr]      = c00[v];
                      outf[(long)r1*ldo + nb + lr]      = c10[v]; }
            if (v1) { outf[(long)r0*ldo + nb + 16 + lr] = c01[v];
                      outf[(long)r1*ldo + nb + 16 + lr] = c11[v]; }
        }
    }
}

// ---------------------------------------------------------------------------
// dense_mix: hs[b,l,:] += sum_{w,k} X[b, l+w-5, k] * Wm[n, w*H+k]
// async-LDS-staged A (37 rows: l0-5 .. l0+31, causal rows zero-filled),
// 2x2 tiles per wave, fused f32 residual. grid=(MTOT/32, 2), block=128.
// ---------------------------------------------------------------------------
#define DMROWS (32 + WWIN - 1)   /* 37 */
__global__ void dense_mix_wmma_f16(const _Float16* __restrict__ X,   // (MTOT,H)
                                   const _Float16* __restrict__ Wm,  // (H, 6H)
                                   float* __restrict__ hs)
{
    __shared__ __align__(16) _Float16 sA[DMROWS*HDIM];

    int tid  = threadIdx.x;
    int lane = tid & 31;
    int wave = tid >> 5;
    int m0 = blockIdx.x * 32;
    int nb = blockIdx.y * 128 + wave*32;
    int hl = lane >> 4;
    int lr = lane & 15;
    int bb = m0 / LSEQ;                 // 3072 % 32 == 0: no batch straddle
    int l0 = m0 % LSEQ;

    // ---- stage rows l0-5 .. l0+31 (zero-fill l<0) ----
    int nch = DMROWS*(HDIM/8);
    const h8 zz = {};
    for (int ci = tid; ci < nch; ci += 128) {
        int row = ci / (HDIM/8);
        int c8  = (ci % (HDIM/8))*8;
        int gl  = l0 - (WWIN-1) + row;
        if (gl >= 0)
            async_copy16(X + ((long)bb*LSEQ + gl)*HDIM + c8, sA + row*HDIM + c8);
        else
            *(h8*)(sA + row*HDIM + c8) = zz;
    }
    async_join();
    __syncthreads();

    v8f c00, c01, c10, c11;
    #pragma unroll
    for (int v = 0; v < 8; v++) {
        int r0 = m0 + v + 8*hl, r1 = r0 + 16;
        c00[v] = hs[(long)r0*HDIM + nb + lr];
        c01[v] = hs[(long)r0*HDIM + nb + 16 + lr];
        c10[v] = hs[(long)r1*HDIM + nb + lr];
        c11[v] = hs[(long)r1*HDIM + nb + 16 + lr];
    }

    const _Float16* w0 = Wm + (long)(nb      + lr)*(WWIN*HDIM);
    const _Float16* w1 = Wm + (long)(nb + 16 + lr)*(WWIN*HDIM);

    for (int w = 0; w < WWIN; w++) {
        // staged-local row for output row r: (r - l0) + w  (offset +5 built in)
        const _Float16* a0 = sA + (lr      + w)*HDIM;
        const _Float16* a1 = sA + (lr + 16 + w)*HDIM;
        for (int k = 0; k < HDIM; k += 32) {
            frag16 fa0, fa1, fb0, fb1;
            fa0.h[0] = *(const h8*)(a0 + k + 8*hl);
            fa0.h[1] = *(const h8*)(a0 + k + 16 + 8*hl);
            fa1.h[0] = *(const h8*)(a1 + k + 8*hl);
            fa1.h[1] = *(const h8*)(a1 + k + 16 + 8*hl);
            fb0.h[0] = *(const h8*)(w0 + w*HDIM + k + 8*hl);
            fb0.h[1] = *(const h8*)(w0 + w*HDIM + k + 16 + 8*hl);
            fb1.h[0] = *(const h8*)(w1 + w*HDIM + k + 8*hl);
            fb1.h[1] = *(const h8*)(w1 + w*HDIM + k + 16 + 8*hl);
            c00 = __builtin_amdgcn_wmma_f32_16x16x32_f16(false, fa0.v, false, fb0.v, (short)0, c00, false, false);
            c10 = __builtin_amdgcn_wmma_f32_16x16x32_f16(false, fa1.v, false, fb0.v, (short)0, c10, false, false);
            c01 = __builtin_amdgcn_wmma_f32_16x16x32_f16(false, fa0.v, false, fb1.v, (short)0, c01, false, false);
            c11 = __builtin_amdgcn_wmma_f32_16x16x32_f16(false, fa1.v, false, fb1.v, (short)0, c11, false, false);
        }
    }

    #pragma unroll
    for (int v = 0; v < 8; v++) {
        int r0 = m0 + v + 8*hl, r1 = r0 + 16;
        hs[(long)r0*HDIM + nb + lr]      = c00[v];
        hs[(long)r0*HDIM + nb + 16 + lr] = c01[v];
        hs[(long)r1*HDIM + nb + lr]      = c10[v];
        hs[(long)r1*HDIM + nb + 16 + lr] = c11[v];
    }
}

// ---------------------------------------------------------------------------
// Causal depthwise conv (DC=4) + SiLU, f16 in / f16 out, f32 math
// ---------------------------------------------------------------------------
__global__ void conv_silu_kernel(const _Float16* __restrict__ xi,
                                 const float* __restrict__ cw,   // (H, DC)
                                 const float* __restrict__ cb,   // (H)
                                 _Float16* __restrict__ xc)
{
    int tok = blockIdx.x;
    int bb  = tok / LSEQ;
    int l   = tok % LSEQ;
    int h   = threadIdx.x;
    float acc = cb[h];
    #pragma unroll
    for (int j = 0; j < DCONV; j++) {
        int ls = l - (DCONV - 1) + j;
        if (ls >= 0)
            acc += (float)xi[((long)bb*LSEQ + ls)*HDIM + h] * cw[h*DCONV + j];
    }
    xc[(long)tok*HDIM + h] = (_Float16)siluf_(acc);
}

// ---------------------------------------------------------------------------
// Selective-scan: grid (BATCH, H/64), block 64 (2 waves -> cheap barriers).
// Fuses dt low-rank GEMM + softplus + SSM recurrence (16 states in regs) +
// D skip + SiLU(z) gate. xdb row (48) double-buffered in LDS, 1 barrier/step.
// ---------------------------------------------------------------------------
__global__ void scan_kernel(const _Float16* __restrict__ u,    // xc   (MTOT,H)
                            const _Float16* __restrict__ xdb,  // (MTOT,48)
                            const _Float16* __restrict__ z,    // (MTOT,H)
                            const float* __restrict__ dtW,  // (H,16)
                            const float* __restrict__ dtb,  // (H)
                            const float* __restrict__ Alog, // (H,16)
                            const float* __restrict__ Dp,   // (H)
                            _Float16* __restrict__ outp)    // gated (MTOT,H)
{
    int bb = blockIdx.x;
    int h  = blockIdx.y*64 + threadIdx.x;
    int ht = threadIdx.x;

    float wdt[DTRANK], Ar[DSTATE], hst[DSTATE];
    #pragma unroll
    for (int r = 0; r < DTRANK; r++) wdt[r] = dtW[h*DTRANK + r];
    #pragma unroll
    for (int s = 0; s < DSTATE; s++) { Ar[s] = -__expf(Alog[h*DSTATE + s]); hst[s] = 0.0f; }
    float dbias = dtb[h];
    float dp    = Dp[h];

    __shared__ float sx[2][48];
    if (ht < 48) sx[0][ht] = (float)xdb[((long)bb*LSEQ)*48 + ht];
    __syncthreads();

    for (int l = 0; l < LSEQ; l++) {
        int p = l & 1;
        if ((l + 1 < LSEQ) && (ht < 48))
            sx[p ^ 1][ht] = (float)xdb[((long)bb*LSEQ + l + 1)*48 + ht];

        float dtraw = dbias;
        #pragma unroll
        for (int r = 0; r < DTRANK; r++) dtraw += sx[p][r]*wdt[r];
        float dt = softplusf_(dtraw);

        long idx = ((long)bb*LSEQ + l)*HDIM + h;
        float ut  = (float)u[idx];
        float dtu = dt*ut;
        float y = 0.0f;
        #pragma unroll
        for (int s = 0; s < DSTATE; s++) {
            hst[s] = __expf(dt*Ar[s])*hst[s] + dtu*sx[p][DTRANK + s];
            y += hst[s]*sx[p][DTRANK + DSTATE + s];
        }
        float zt = (float)z[idx];
        outp[idx] = (_Float16)((y + ut*dp) * siluf_(zt));
        __syncthreads();
    }
}

// ---------------------------------------------------------------------------
// Final: LN(state tokens, fp32) -> policy head -> tanh. One block per (b,t).
// ---------------------------------------------------------------------------
__global__ void final_kernel(const float* __restrict__ hs,
                             const float* __restrict__ gf, const float* __restrict__ bfv,
                             const float* __restrict__ paW, const float* __restrict__ pab,
                             float* __restrict__ outp)
{
    int bt = blockIdx.x;                 // 0..B*T-1
    int bb = bt / TSEQ;
    int t  = bt % TSEQ;
    int h  = threadIdx.x;
    long row = ((long)bb*LSEQ + 3*t + 1)*HDIM;   // state token
    float x = hs[row + h];
    __shared__ float s1[HDIM], s2[HDIM], lnv[HDIM];
    s1[h] = x; s2[h] = x*x;
    __syncthreads();
    for (int off = 128; off > 0; off >>= 1) {
        if (h < off) { s1[h] += s1[h+off]; s2[h] += s2[h+off]; }
        __syncthreads();
    }
    float mean = s1[0] * (1.0f/HDIM);
    float var  = s2[0] * (1.0f/HDIM) - mean*mean;
    float rstd = rsqrtf(var + EPSF);
    lnv[h] = (x - mean)*rstd*gf[h] + bfv[h];
    __syncthreads();
    if (h < ADIM) {
        float acc = pab[h];
        for (int k = 0; k < HDIM; k++) acc += lnv[k]*paW[h*HDIM + k];
        outp[(long)bt*ADIM + h] = tanhf(acc);
    }
}

// ---------------------------------------------------------------------------
extern "C" void kernel_launch(void* const* d_in, const int* in_sizes, int n_in,
                              void* d_out, int out_size, void* d_ws, size_t ws_size,
                              hipStream_t stream)
{
    const float* states  = (const float*)d_in[0];
    const float* actions = (const float*)d_in[1];
    const float* rtg     = (const float*)d_in[2];
    const float* W_ret   = (const float*)d_in[3];
    const float* b_ret   = (const float*)d_in[4];
    const float* W_st    = (const float*)d_in[5];
    const float* b_st    = (const float*)d_in[6];
    const float* W_act   = (const float*)d_in[7];
    const float* b_act   = (const float*)d_in[8];
    const float* g_emb   = (const float*)d_in[9];
    const float* be_emb  = (const float*)d_in[10];
    const float* g1      = (const float*)d_in[11];
    const float* b1      = (const float*)d_in[12];
    const float* g2      = (const float*)d_in[13];
    const float* b2      = (const float*)d_in[14];
    const float* gf      = (const float*)d_in[15];
    const float* bfv     = (const float*)d_in[16];
    const float* mix_W   = (const float*)d_in[17];
    const float* in_W    = (const float*)d_in[18];
    const float* conv_w  = (const float*)d_in[19];
    const float* conv_b  = (const float*)d_in[20];
    const float* xproj_W = (const float*)d_in[21];
    const float* dt_W    = (const float*)d_in[22];
    const float* dt_b    = (const float*)d_in[23];
    const float* A_log   = (const float*)d_in[24];
    const float* D_p     = (const float*)d_in[25];
    const float* out_W   = (const float*)d_in[26];
    const float* pa_W    = (const float*)d_in[27];
    const float* pa_b    = (const float*)d_in[28];
    float* outp = (float*)d_out;
    (void)in_sizes; (void)n_in; (void)out_size; (void)ws_size;

    // workspace layout
    const size_t SBUF = (size_t)MTOT * HDIM;           // 25,165,824 elems
    float*    hs = (float*)d_ws;                       // f32 residual stream
    _Float16* h0 = (_Float16*)(hs + SBUF);             // LN out, then xc
    _Float16* h1 = h0 + SBUF;                          // xi, then gated scan out
    _Float16* h2 = h1 + SBUF;                          // z
    _Float16* h4 = h2 + SBUF;                          // xdb (MTOT,48)
    _Float16* wmix16 = h4 + (size_t)MTOT*48;           // NL*H*6H
    _Float16* win16  = wmix16 + (size_t)NLAYERS*HDIM*WWIN*HDIM;
    _Float16* wxp16  = win16  + (size_t)NLAYERS*2*HDIM*HDIM;
    _Float16* wout16 = wxp16  + (size_t)NLAYERS*(DTRANK+2*DSTATE)*HDIM;

    dim3 blk256(256), blk128(128), blk64(64);

    // weight downcast (tiny; resident in L2 for all M-tiles)
    {
        int nmix = NLAYERS*HDIM*WWIN*HDIM;
        int nin  = NLAYERS*2*HDIM*HDIM;
        int nxp  = NLAYERS*(DTRANK+2*DSTATE)*HDIM;
        int nout = NLAYERS*HDIM*HDIM;
        f32_to_f16_kernel<<<(nmix+255)/256, blk256, 0, stream>>>(mix_W,   wmix16, nmix);
        f32_to_f16_kernel<<<(nin +255)/256, blk256, 0, stream>>>(in_W,    win16,  nin);
        f32_to_f16_kernel<<<(nxp +255)/256, blk256, 0, stream>>>(xproj_W, wxp16,  nxp);
        f32_to_f16_kernel<<<(nout+255)/256, blk256, 0, stream>>>(out_W,   wout16, nout);
    }

    embed_ln_kernel<<<MTOT, blk256, 0, stream>>>(states, actions, rtg,
        W_ret, b_ret, W_st, b_st, W_act, b_act, g_emb, be_emb, hs);

    for (int i = 0; i < NLAYERS; i++) {
        const _Float16* mW  = wmix16 + (size_t)i*HDIM*WWIN*HDIM;
        const _Float16* iW  = win16  + (size_t)i*2*HDIM*HDIM;
        const _Float16* xpW = wxp16  + (size_t)i*(DTRANK+2*DSTATE)*HDIM;
        const _Float16* oW  = wout16 + (size_t)i*HDIM*HDIM;
        const float* cw  = conv_w + (size_t)i*HDIM*DCONV;
        const float* cb  = conv_b + (size_t)i*HDIM;
        const float* dW  = dt_W   + (size_t)i*HDIM*DTRANK;
        const float* db  = dt_b   + (size_t)i*HDIM;
        const float* Al  = A_log  + (size_t)i*HDIM*DSTATE;
        const float* Dpp = D_p    + (size_t)i*HDIM;

        // hs += dense_mix(LN1(hs))
        ln_f16_kernel<<<MTOT, blk256, 0, stream>>>(hs, g1, b1, h0);
        dense_mix_wmma_f16<<<dim3(MTOT/32, 2), blk128, 0, stream>>>(h0, mW, hs);

        // mamba
        ln_f16_kernel<<<MTOT, blk256, 0, stream>>>(hs, g2, b2, h0);
        gemm_f16_wmma<<<dim3(MTOT/32, 2), blk128, 0, stream>>>(
            h0, HDIM, iW,             HDIM, HDIM, nullptr, h1, HDIM, HDIM, 0); // xi (f16)
        gemm_f16_wmma<<<dim3(MTOT/32, 2), blk128, 0, stream>>>(
            h0, HDIM, iW + HDIM*HDIM, HDIM, HDIM, nullptr, h2, HDIM, HDIM, 0); // z (f16)
        conv_silu_kernel<<<MTOT, blk256, 0, stream>>>(h1, cw, cb, h0);          // xc (f16)
        gemm_f16_wmma<<<dim3(MTOT/32, 1), blk128, 0, stream>>>(
            h0, HDIM, xpW, HDIM, HDIM, nullptr, h4, 48, 48, 0);                 // xdb (f16)
        scan_kernel<<<dim3(BATCH, HDIM/64), blk64, 0, stream>>>(
            h0, h4, h2, dW, db, Al, Dpp, h1);
        gemm_f16_wmma<<<dim3(MTOT/32, 2), blk128, 0, stream>>>(
            h1, HDIM, oW, HDIM, HDIM, hs, nullptr, HDIM, HDIM, 1);              // hs += (f32)
    }

    final_kernel<<<BATCH*TSEQ, blk256, 0, stream>>>(hs, gf, bfv, pa_W, pa_b, outp);
}
